// DeeperGCN_24232205484640
// MI455X (gfx1250) — matrix operand order repaired and checked by reading, not compiled, not run on hardware
//
#include <hip/hip_runtime.h>
#include <hip/hip_bf16.h>

typedef _Float16 half_t;
typedef __attribute__((ext_vector_type(16))) _Float16 v16h;
typedef __attribute__((ext_vector_type(8)))  _Float16 v8h;
typedef __attribute__((ext_vector_type(8)))  float    v8f;

#define DEVI __device__ __forceinline__

constexpr int   L_ = 20;
constexpr int   D_ = 128;
constexpr int   H_ = 256;
constexpr int   N_ = 100000;   // divisible by 32 -> 3125 GEMM row blocks
constexpr int   E_ = 400000;
constexpr int   G_ = 2048;
constexpr float MSG_EPS_ = 1e-7f;
constexpr float BN_EPS_  = 1e-5f;

// ---------------- helpers ----------------

DEVI unsigned enc_f32(float x) {           // order-preserving float->uint map
  unsigned u = __float_as_uint(x);
  return (u & 0x80000000u) ? ~u : (u | 0x80000000u);
}
DEVI float dec_f32(unsigned u) {
  unsigned v = (u & 0x80000000u) ? (u & 0x7FFFFFFFu) : ~u;
  return __uint_as_float(v);
}

// ---------------- utility kernels ----------------

__global__ void fill_u32_kernel(unsigned* p, unsigned v, long n) {
  long i = (long)blockIdx.x * blockDim.x + threadIdx.x;
  long s = (long)gridDim.x * blockDim.x;
  for (; i < n; i += s) p[i] = v;
}

struct Ptr9 { const float* p[9]; };

// h[n,c] = sum_i atom_emb[i][x[n,i], c]
__global__ void atom_kernel(const int* __restrict__ x, Ptr9 emb, float* __restrict__ h) {
  int n = blockIdx.x;
  int c = threadIdx.x;
  const int* xr = x + (long)n * 9;
  float s = 0.f;
#pragma unroll
  for (int i = 0; i < 9; ++i) s += emb.p[i][(long)xr[i] * D_ + c];
  h[(long)n * D_ + c] = s;
}

// convert [layers][K][M] f32 weights -> [layers][M][K] f16 (transposed per layer)
__global__ void wcvt_kernel(const float* __restrict__ W, half_t* __restrict__ Wt,
                            int K, int M, long total) {
  long i = (long)blockIdx.x * blockDim.x + threadIdx.x;
  long s = (long)gridDim.x * blockDim.x;
  long km = (long)K * M;
  for (; i < total; i += s) {
    long l = i / km, rem = i % km;
    long k = rem / M, m = rem % M;
    Wt[(l * M + m) * K + k] = (half_t)W[i];
  }
}

// column sums / sumsq for training-mode BN (over node axis)
__global__ void colstat_kernel(const float* __restrict__ x, float* __restrict__ sum,
                               float* __restrict__ sq, int ncols, int nrows) {
  int tid = blockIdx.x * blockDim.x + threadIdx.x;
  int nth = gridDim.x * blockDim.x;      // multiple of ncols by construction
  int c = tid % ncols;
  int r0 = tid / ncols;
  int rs = nth / ncols;
  float s = 0.f, q = 0.f;
  for (int r = r0; r < nrows; r += rs) {
    float v = x[(long)r * ncols + c];
    s += v; q += v * v;
  }
  atomicAdd(&sum[c], s);
  atomicAdd(&sq[c], q);
}

__global__ void statfin_kernel(const float* sum, const float* sq,
                               const float* g, const float* b,
                               float* sc, float* sh, int ncols, float invn) {
  int c = blockIdx.x * blockDim.x + threadIdx.x;
  if (c >= ncols) return;
  float mu  = sum[c] * invn;
  float var = fmaxf(sq[c] * invn - mu * mu, 0.f);
  float s   = g[c] * rsqrtf(var + BN_EPS_);
  sc[c] = s;
  sh[c] = b[c] - mu * s;
}

__global__ void nrelu_f32_kernel(const float* __restrict__ x, const float* __restrict__ sc,
                                 const float* __restrict__ sh, float* __restrict__ y,
                                 int ncols, long total) {
  long i = (long)blockIdx.x * blockDim.x + threadIdx.x;
  if (i >= total) return;
  int c = (int)(i % ncols);
  y[i] = fmaxf(x[i] * sc[c] + sh[c], 0.f);
}

__global__ void nrelu_f16_kernel(const float* __restrict__ x, const float* __restrict__ sc,
                                 const float* __restrict__ sh, half_t* __restrict__ y,
                                 int ncols, long total) {
  long i = (long)blockIdx.x * blockDim.x + threadIdx.x;
  if (i >= total) return;
  int c = (int)(i % ncols);
  y[i] = (half_t)fmaxf(x[i] * sc[c] + sh[c], 0.f);
}

// ---------------- edge phase (scatter softmax aggregation) ----------------

__global__ void edge_max_kernel(const float* __restrict__ h1, const int* __restrict__ src,
                                const int* __restrict__ dst, const int* __restrict__ ea,
                                const float* __restrict__ b0, const float* __restrict__ b1,
                                const float* __restrict__ b2, const float* __restrict__ tp,
                                unsigned* __restrict__ mxu) {
  int e = blockIdx.x * 2 + (threadIdx.x >> 7);
  int c = threadIdx.x & 127;
  if (e >= E_) return;
  float t = tp[0];
  int s = src[e], d = dst[e];
  const int* er = ea + (long)e * 3;
  float eav = b0[(long)er[0] * D_ + c] + b1[(long)er[1] * D_ + c] + b2[(long)er[2] * D_ + c];
  float m = h1[(long)s * D_ + c] + eav;
  m = fmaxf(m, 0.f) + MSG_EPS_;
  atomicMax(mxu + (long)d * D_ + c, enc_f32(m * t));
}

// den[dst] += exp(st-mx);  wnum[dst] += msg*exp(st-mx)  (alpha division folded out)
__global__ void edge_sum_kernel(const float* __restrict__ h1, const int* __restrict__ src,
                                const int* __restrict__ dst, const int* __restrict__ ea,
                                const float* __restrict__ b0, const float* __restrict__ b1,
                                const float* __restrict__ b2, const float* __restrict__ tp,
                                const unsigned* __restrict__ mxu,
                                float* __restrict__ den, float* __restrict__ wnum) {
  int e = blockIdx.x * 2 + (threadIdx.x >> 7);
  int c = threadIdx.x & 127;
  if (e >= E_) return;
  float t = tp[0];
  int s = src[e], d = dst[e];
  const int* er = ea + (long)e * 3;
  float eav = b0[(long)er[0] * D_ + c] + b1[(long)er[1] * D_ + c] + b2[(long)er[2] * D_ + c];
  float m = h1[(long)s * D_ + c] + eav;
  m = fmaxf(m, 0.f) + MSG_EPS_;
  float mx = dec_f32(mxu[(long)d * D_ + c]);
  float ex = __expf(m * t - mx);
  atomicAdd(&den[(long)d * D_ + c], ex);
  atomicAdd(&wnum[(long)d * D_ + c], m * ex);
}

// z = h1 + wnum/den  -> f16 GEMM input
__global__ void make_z16_kernel(const float* __restrict__ h1, const float* __restrict__ wnum,
                                const float* __restrict__ den, half_t* __restrict__ z16,
                                long total) {
  long i = (long)blockIdx.x * blockDim.x + threadIdx.x;
  if (i >= total) return;
  float dn = den[i];
  float m = dn > 0.f ? wnum[i] / dn : 0.f;
  z16[i] = (half_t)(h1[i] + m);
}

// ---------------- WMMA GEMM ----------------
// A: [N x K] f16 row-major. Bt: [M x K] f16 (weights pre-transposed).
// Block: 256 threads = 8 waves, BM=32 rows (2x 16-row tiles), BN=M.
// Wave w owns column tiles {w, w+8} (if M==256) or {w} (if M==128).
// FUSE: C = acc + bias[col] + resid (GEMM3: bias=b3, resid=h for residual layers).

DEVI v16h load_a_frag(const half_t* rowbase, int lh) {
  // A 16x32 frag, k order chosen here (same as B): lane<16 -> K 0..15, lane>=16 -> K 16..31 (contiguous 32B)
  return *(const v16h*)(rowbase + lh * 16);
}

template <int K, int M, bool FUSE>
__global__ void __launch_bounds__(256)
gemm_wmma_kernel(const half_t* __restrict__ A, const half_t* __restrict__ Bt,
                 const float* __restrict__ bias, const float* __restrict__ resid,
                 float* __restrict__ C) {
  constexpr int CPW = M / 128;            // column tiles per wave (1 or 2)
  const int wave = threadIdx.x >> 5;
  const int lane = threadIdx.x & 31;
  const int lm = lane & 15;
  const int lh = lane >> 4;
  const long row0 = (long)blockIdx.x * 32;

  v8f acc[2][CPW];
#pragma unroll
  for (int rt = 0; rt < 2; ++rt)
#pragma unroll
    for (int ct = 0; ct < CPW; ++ct)
      acc[rt][ct] = (v8f){0.f, 0.f, 0.f, 0.f, 0.f, 0.f, 0.f, 0.f};

  const half_t* arow0 = A + (row0 + lm) * K;
  const half_t* arow1 = arow0 + (long)16 * K;

#pragma unroll 2
  for (int kk = 0; kk < K; kk += 32) {
    v16h af0 = load_a_frag(arow0 + kk, lh);
    v16h af1 = load_a_frag(arow1 + kk, lh);
#pragma unroll
    for (int ct = 0; ct < CPW; ++ct) {
      const int col = (wave + ct * 8) * 16 + lm;
      // B 32x16 frag, same k order as A: lanes 0-15 K=0..15, lanes 16-31 K=16..31 (contiguous 32B)
      v16h bf = *(const v16h*)(Bt + (long)col * K + kk + lh * 16);
      acc[0][ct] = __builtin_amdgcn_wmma_f32_16x16x32_f16(
          false, af0, false, bf, (short)0, acc[0][ct], false, false);
      acc[1][ct] = __builtin_amdgcn_wmma_f32_16x16x32_f16(
          false, af1, false, bf, (short)0, acc[1][ct], false, false);
    }
  }

  // C/D layout: VGPR v, lanes 0-15 -> row v, lanes 16-31 -> row v+8, col = lane&15
#pragma unroll
  for (int rt = 0; rt < 2; ++rt) {
    const long rbase = row0 + rt * 16 + lh * 8;
#pragma unroll
    for (int ct = 0; ct < CPW; ++ct) {
      const int col = (wave + ct * 8) * 16 + lm;
      if constexpr (FUSE) {
        const float badd = bias[col];
        const bool has_res = (resid != nullptr);
#pragma unroll
        for (int v = 0; v < 8; ++v) {
          long idx = (rbase + v) * M + col;
          float val = acc[rt][ct][v] + badd;
          if (has_res) val += resid[idx];
          C[idx] = val;
        }
      } else {
#pragma unroll
        for (int v = 0; v < 8; ++v) {
          long idx = (rbase + v) * M + col;
          C[idx] = acc[rt][ct][v];        // no +0.f: avoids dead v_pk_add_f32 ops
        }
      }
    }
  }
}

// ---------------- pooling + readout ----------------

__global__ void pool_kernel(const float* __restrict__ hf, const int* __restrict__ batch,
                            float* __restrict__ hs, float* __restrict__ cnt) {
  int n = blockIdx.x;
  int c = threadIdx.x;
  int g = batch[n];
  atomicAdd(&hs[(long)g * D_ + c], hf[(long)n * D_ + c]);
  if (c == 0) atomicAdd(&cnt[g], 1.0f);
}

__global__ void out_kernel(const float* __restrict__ hs, const float* __restrict__ cnt,
                           const float* __restrict__ Wo, const float* __restrict__ bo,
                           float* __restrict__ out) {
  int g = blockIdx.x * blockDim.x + threadIdx.x;
  if (g >= G_) return;
  float inv = 1.f / fmaxf(cnt[g], 1.f);
  float acc = bo[0];
#pragma unroll 4
  for (int c = 0; c < D_; ++c) acc += hs[(long)g * D_ + c] * inv * Wo[c];
  out[g] = acc;
}

// ---------------- orchestration ----------------

extern "C" void kernel_launch(void* const* d_in, const int* in_sizes, int n_in,
                              void* d_out, int out_size, void* d_ws, size_t ws_size,
                              hipStream_t stream) {
  if (n_in < 31) return;
  // params (setup_inputs dict insertion order, recursively flattened)
  Ptr9 emb;
  for (int i = 0; i < 9; ++i) emb.p[i] = (const float*)d_in[i];
  const float* bond0 = (const float*)d_in[9];    // [L,5,128]
  const float* bond1 = (const float*)d_in[10];   // [L,6,128]
  const float* bond2 = (const float*)d_in[11];   // [L,2,128]
  const float* W1  = (const float*)d_in[12];     // [L,128,256]  (b1 at 13 absorbed by BN)
  const float* g1  = (const float*)d_in[14];
  const float* be1 = (const float*)d_in[15];
  const float* W2  = (const float*)d_in[16];     // [L,256,256]  (b2 at 17 absorbed by BN)
  const float* g2  = (const float*)d_in[18];
  const float* be2 = (const float*)d_in[19];
  const float* W3  = (const float*)d_in[20];     // [L,256,128]
  const float* b3  = (const float*)d_in[21];
  const float* tpar= (const float*)d_in[22];     // [L]
  const float* ng  = (const float*)d_in[23];
  const float* nb  = (const float*)d_in[24];
  const float* Wo  = (const float*)d_in[25];
  const float* bo  = (const float*)d_in[26];
  const int* x     = (const int*)d_in[27];       // [N,9]
  const int* eattr = (const int*)d_in[28];       // [E,3]
  const int* eidx  = (const int*)d_in[29];       // [2,E]
  const int* batch = (const int*)d_in[30];       // [N]
  const int* src = eidx;
  const int* dst = eidx + E_;

  // workspace carve (~286 MB)
  char* base = (char*)d_ws;
  size_t off = 0;
  auto carve = [&](size_t bytes) -> void* {
    void* p = base + off;
    off += (bytes + 255) & ~(size_t)255;
    return p;
  };
  float*  h    = (float*)  carve((size_t)N_ * D_ * 4);
  float*  h1   = (float*)  carve((size_t)N_ * D_ * 4);
  void*   wa   =           carve((size_t)N_ * H_ * 2);  // wnum f32[N*D] | a16 f16[N*H]
  void*   ra   =           carve((size_t)N_ * H_ * 4);  // {mxu,den} f32[N*D]x2 | y f32[N*H]
  half_t* z16  = (half_t*) carve((size_t)N_ * D_ * 2);
  half_t* w1t  = (half_t*) carve((size_t)L_ * D_ * H_ * 2);
  half_t* w2t  = (half_t*) carve((size_t)L_ * H_ * H_ * 2);
  half_t* w3t  = (half_t*) carve((size_t)L_ * H_ * D_ * 2);
  float*  stat = (float*)  carve(1024 * 4);
  float*  hs   = (float*)  carve(((size_t)G_ * D_ + G_) * 4);
  if (off > ws_size) return;

  float*    wnum = (float*)wa;
  half_t*   a16  = (half_t*)wa;
  unsigned* mxu  = (unsigned*)ra;
  float*    den  = (float*)ra + (size_t)N_ * D_;
  float*    y    = (float*)ra;
  float* stat_sum = stat, *stat_sq = stat + 256, *stat_sc = stat + 512, *stat_sh = stat + 768;
  float* cnt = hs + (size_t)G_ * D_;

  const long ND = (long)N_ * D_;
  const long NH = (long)N_ * H_;
  const int  TB = 256;
  auto blks = [](long n) { return (int)((n + 255) / 256); };

  auto bn_stats = [&](const float* xbuf, const float* g, const float* b, int ncols) {
    fill_u32_kernel<<<4, TB, 0, stream>>>((unsigned*)stat_sum, 0u, 512);
    colstat_kernel<<<256, TB, 0, stream>>>(xbuf, stat_sum, stat_sq, ncols, N_);
    statfin_kernel<<<1, TB, 0, stream>>>(stat_sum, stat_sq, g, b, stat_sc, stat_sh,
                                         ncols, 1.f / (float)N_);
  };

  // weights -> f16 transposed [M][K] per layer (done each call; idempotent)
  wcvt_kernel<<<2048, TB, 0, stream>>>(W1, w1t, D_, H_, (long)L_ * D_ * H_);
  wcvt_kernel<<<2048, TB, 0, stream>>>(W2, w2t, H_, H_, (long)L_ * H_ * H_);
  wcvt_kernel<<<2048, TB, 0, stream>>>(W3, w3t, H_, D_, (long)L_ * H_ * D_);

  // AtomEncoder
  atom_kernel<<<N_, D_, 0, stream>>>(x, emb, h);

  for (int l = 0; l < L_; ++l) {
    // pre-norm (layers 1..L-1): h1 = relu(bn(h)); layer 0 uses h directly
    const float* h1p;
    if (l > 0) {
      bn_stats(h, ng + (long)(l - 1) * D_, nb + (long)(l - 1) * D_, D_);
      nrelu_f32_kernel<<<blks(ND), TB, 0, stream>>>(h, stat_sc, stat_sh, h1, D_, ND);
      h1p = h1;
    } else {
      h1p = h;
    }

    // edge phase: scatter softmax aggregation
    const float* be0 = bond0 + (long)l * 5 * D_;
    const float* be1b = bond1 + (long)l * 6 * D_;
    const float* be2b = bond2 + (long)l * 2 * D_;
    const float* tl = tpar + l;
    fill_u32_kernel<<<2048, TB, 0, stream>>>((unsigned*)ra, 0u, NH);   // mxu + den
    fill_u32_kernel<<<2048, TB, 0, stream>>>((unsigned*)wnum, 0u, ND);
    edge_max_kernel<<<E_ / 2, TB, 0, stream>>>(h1p, src, dst, eattr, be0, be1b, be2b, tl, mxu);
    edge_sum_kernel<<<E_ / 2, TB, 0, stream>>>(h1p, src, dst, eattr, be0, be1b, be2b, tl,
                                               mxu, den, wnum);
    make_z16_kernel<<<blks(ND), TB, 0, stream>>>(h1p, wnum, den, z16, ND);

    // MLP: 128 -> 256 (BN+ReLU) -> 256 (BN+ReLU) -> 128 (+b3, +residual)
    gemm_wmma_kernel<128, 256, false><<<N_ / 32, TB, 0, stream>>>(
        z16, w1t + (long)l * H_ * D_, nullptr, nullptr, y);
    bn_stats(y, g1 + (long)l * H_, be1 + (long)l * H_, H_);
    nrelu_f16_kernel<<<blks(NH), TB, 0, stream>>>(y, stat_sc, stat_sh, a16, H_, NH);

    gemm_wmma_kernel<256, 256, false><<<N_ / 32, TB, 0, stream>>>(
        a16, w2t + (long)l * H_ * H_, nullptr, nullptr, y);
    bn_stats(y, g2 + (long)l * H_, be2 + (long)l * H_, H_);
    nrelu_f16_kernel<<<blks(NH), TB, 0, stream>>>(y, stat_sc, stat_sh, a16, H_, NH);

    gemm_wmma_kernel<256, 128, true><<<N_ / 32, TB, 0, stream>>>(
        a16, w3t + (long)l * H_ * D_, b3 + (long)l * D_,
        (l > 0) ? h : nullptr, h);                      // h = MLP(z) (+ h if residual)
  }

  // final norm + relu, mean pool per graph, linear readout
  bn_stats(h, ng + (long)(L_ - 1) * D_, nb + (long)(L_ - 1) * D_, D_);
  nrelu_f32_kernel<<<blks(ND), TB, 0, stream>>>(h, stat_sc, stat_sh, h1, D_, ND);
  fill_u32_kernel<<<256, TB, 0, stream>>>((unsigned*)hs, 0u, (long)G_ * D_ + G_);
  pool_kernel<<<N_, D_, 0, stream>>>(h1, batch, hs, cnt);
  out_kernel<<<(G_ + TB - 1) / TB, TB, 0, stream>>>(hs, cnt, Wo, bo, (float*)d_out);
}
